// GRUDecoder_30726196035717
// MI455X (gfx1250) — matrix operand order repaired and checked by reading, not compiled, not run on hardware
//
#include <hip/hip_runtime.h>

typedef __attribute__((ext_vector_type(16))) _Float16 v16h;
typedef __attribute__((ext_vector_type(8)))  float    v8f;

static constexpr int HDIM = 32;             // hidden dim (fixed by problem)
static constexpr int WAVES_PER_BLOCK = 4;   // wave32 -> 128 threads/block
static constexpr int ROWS_PER_WAVE = 16;    // one 16-row batch tile per wave
static constexpr int ROWS_PER_BLOCK = WAVES_PER_BLOCK * ROWS_PER_WAVE;

#if defined(__has_builtin)
#if __has_builtin(__builtin_amdgcn_permlane16_swap)
#define USE_PL16_SWAP 1
#endif
#endif

// Fast transcendentals: single v_exp_f32 / v_rcp_f32 instead of the
// ~10-instruction IEEE division expansion.  These are TRANS ops and can
// co-execute with the WMMA pipe.
__device__ __forceinline__ float sigmoid_f(float x) {
  return __builtin_amdgcn_rcpf(1.0f + __expf(-x));
}
__device__ __forceinline__ float tanh_f(float x) {
  float t = __expf(-2.0f * x);
  return (1.0f - t) * __builtin_amdgcn_rcpf(1.0f + t);
}

// Pack two f32 into one dword of two f16 (v_cvt_pkrtz_f16_f32, 1 instr).
__device__ __forceinline__ unsigned pk16(float a, float b) {
  return __builtin_bit_cast(unsigned, __builtin_amdgcn_cvt_pkrtz(a, b));
}

// Load a 16x32 f16 A-fragment (tile of a [96,32] row-major f32 weight matrix).
// A k order used here (contiguous, matches B): lane<16 holds row M=lane, K=0..15;
// lane>=16 holds row M=lane-16, K=16..31.
__device__ __forceinline__ v16h load_A_tile(const float* __restrict__ W,
                                            int tile, int lane) {
  const int row = tile * 16 + (lane & 15);
  const int kb = (lane < 16) ? 0 : 16;
  const float* p = W + row * HDIM;
  v16h a;
#pragma unroll
  for (int e = 0; e < 8; ++e) a[e] = (_Float16)p[kb + e];
#pragma unroll
  for (int e = 0; e < 8; ++e) a[8 + e] = (_Float16)p[kb + 8 + e];
  return a;
}

// Build B-fragment (32x16 f16, K=hidden, N=batch-col) from the two f32
// C/D-layout h.T tiles.  Convert to packed f16 (v_cvt_pkrtz_f16_f32), then
// move the lane^16 halves into place.
//
// Fast path: v_permlane16_swap_b32 swaps the odd 16-lane row of operand 0
// with the even row of operand 1, so ONE instruction yields both
//   b.u[j]   = {c0.row0 | c1.row0}   (K=2j / K=16+2j)
//   b.u[4+j] = {c0.row1 | c1.row1}   (K=8+2j / K=24+2j)
// -> 8 converts + 4 swaps per step, no LDS traffic on the critical path.
__device__ __forceinline__ v16h make_b_from_h(const v8f& ht0, const v8f& ht1,
                                              bool lo) {
  union { unsigned u[8]; v16h h; } b;
#if defined(USE_PL16_SWAP)
#pragma unroll
  for (int j = 0; j < 4; ++j) {
    unsigned c0 = pk16(ht0[2 * j], ht0[2 * j + 1]);
    unsigned c1 = pk16(ht1[2 * j], ht1[2 * j + 1]);
    auto r = __builtin_amdgcn_permlane16_swap(c0, c1, false, false);
    b.u[j]     = r[0];
    b.u[4 + j] = r[1];
  }
#else
  unsigned c0[4], c1[4];
#pragma unroll
  for (int j = 0; j < 4; ++j) {
    c0[j] = pk16(ht0[2 * j], ht0[2 * j + 1]);
    c1[j] = pk16(ht1[2 * j], ht1[2 * j + 1]);
  }
#pragma unroll
  for (int j = 0; j < 4; ++j) {
    unsigned s0 = (unsigned)__shfl_xor((int)c0[j], 16, 32);
    unsigned s1 = (unsigned)__shfl_xor((int)c1[j], 16, 32);
    b.u[j]     = lo ? c0[j] : s1;   // K =  2j..2j+1 (lo) | 16+2j.. (hi)
    b.u[4 + j] = lo ? s0 : c1[j];   // K = 8+2j..    (lo) | 24+2j.. (hi)
  }
#endif
  return b.h;
}

__device__ __forceinline__ v8f wmma_f16(v16h a, v16h b, v8f c) {
  return __builtin_amdgcn_wmma_f32_16x16x32_f16(false, a, false, b,
                                                (short)0, c, false, false);
}

__global__ __launch_bounds__(WAVES_PER_BLOCK * 32)
void gru_decoder_kernel(const float* __restrict__ in_data,
                        const float* __restrict__ hidden,
                        const float* __restrict__ w_ih,
                        const float* __restrict__ w_hh,
                        const float* __restrict__ b_ih,
                        const float* __restrict__ b_hh,
                        const float* __restrict__ w_out,
                        const float* __restrict__ b_out,
                        const int* __restrict__ nsteps_p,
                        float* __restrict__ out) {
  // Per-k constants, 8 floats (= two float4, 16B aligned -> ds_load_b128):
  // [0..2] rowsum(w_ih) for gates r,z,n (y-feedback:  gx = y*rowsum + b_ih)
  // [3]    b_ih[r]+b_hh[r]   [4] b_ih[z]+b_hh[z]
  // [5]    b_ih[n]           [6] b_hh[n]  (torch: n = tanh(gx_n + r*(gh_n+b_hn)))
  // [7]    w_out[k]
  __shared__ __align__(16) float kc[32][8];
  const int tid = threadIdx.x;
  if (tid < 32) {
    const int k = tid;
    float r0 = 0.f, r1 = 0.f, r2 = 0.f;
#pragma unroll
    for (int i = 0; i < HDIM; ++i) {
      r0 += w_ih[k * HDIM + i];
      r1 += w_ih[(k + 32) * HDIM + i];
      r2 += w_ih[(k + 64) * HDIM + i];
    }
    kc[k][0] = r0; kc[k][1] = r1; kc[k][2] = r2;
    kc[k][3] = b_ih[k] + b_hh[k];
    kc[k][4] = b_ih[k + 32] + b_hh[k + 32];
    kc[k][5] = b_ih[k + 64];
    kc[k][6] = b_hh[k + 64];
    kc[k][7] = w_out[k];
  }
  __syncthreads();

  const int lane = tid & 31;
  const bool lo = lane < 16;
  const int wv = tid >> 5;
  const int col = blockIdx.x * ROWS_PER_BLOCK + wv * 16 + (lane & 15);
  const int nsteps = *nsteps_p;
  const float bo = *b_out;
  const int ko = lo ? 0 : 8;       // C/D layout: lane>=16 holds rows M=8..15
  const int hioff = lo ? 0 : 8;

  // h.T as two f32 C/D-layout tiles: ht0 = k 0..15, ht1 = k 16..31 (cols=batch)
  v8f ht0, ht1;
#pragma unroll
  for (int v = 0; v < 8; ++v) {
    ht0[v] = hidden[col * HDIM + ko + v];
    ht1[v] = hidden[col * HDIM + 16 + ko + v];
  }

  // w_hh A-fragments, resident for all 128 steps (6 x 16x32 f16 = 48 VGPRs)
  v16h wa[6];
#pragma unroll
  for (int t = 0; t < 6; ++t) wa[t] = load_A_tile(w_hh, t, lane);

  const v8f zero8 = {0.f, 0.f, 0.f, 0.f, 0.f, 0.f, 0.f, 0.f};
  float y = 0.0f;

  // ---------------- step 0: real input x = in_data ----------------
  {
    v16h bx;
    const int kb = lo ? 0 : 16;
#pragma unroll
    for (int e = 0; e < 16; ++e)
      bx[e] = (_Float16)in_data[col * HDIM + kb + e];
    v16h bh = make_b_from_h(ht0, ht1, lo);

    v8f Drz[4], Dxn[2], Dhn[2];
#pragma unroll
    for (int t = 0; t < 4; ++t) {           // r,z gates: gx+gh fused via C
      v16h wia = load_A_tile(w_ih, t, lane);
      Drz[t] = wmma_f16(wia, bx, wmma_f16(wa[t], bh, zero8));
    }
#pragma unroll
    for (int t = 0; t < 2; ++t) {           // n gate: keep gx_n, gh_n separate
      v16h wia = load_A_tile(w_ih, 4 + t, lane);
      Dxn[t] = wmma_f16(wia, bx, zero8);
      Dhn[t] = wmma_f16(wa[4 + t], bh, zero8);
    }
    float yp = 0.0f;
#pragma unroll
    for (int t = 0; t < 2; ++t) {
      v8f& hh = (t == 0) ? ht0 : ht1;
#pragma unroll
      for (int v = 0; v < 8; ++v) {
        const float4* cp = (const float4*)kc[t * 16 + v + hioff];
        const float4 ca = cp[0];
        const float4 cb = cp[1];
        float r = sigmoid_f(Drz[t][v] + ca.w);
        float z = sigmoid_f(Drz[2 + t][v] + cb.x);
        float n = tanh_f(Dxn[t][v] + cb.y + r * (Dhn[t][v] + cb.z));
        float hn = (1.0f - z) * n + z * hh[v];
        hh[v] = hn;
        yp = fmaf(hn, cb.w, yp);
      }
    }
    y = yp + __shfl_xor(yp, 16, 32) + bo;   // each lane holds 16 of 32 k's
    if (lo) out[col * nsteps + 0] = y;
  }

  // ---------------- steps 1..N-1: x is broadcast(y) => gx = y*rowsum + b ----
  for (int s = 1; s < nsteps; ++s) {
    v16h bh = make_b_from_h(ht0, ht1, lo);
    v8f D[6];
#pragma unroll
    for (int t = 0; t < 6; ++t) D[t] = wmma_f16(wa[t], bh, zero8);

    float yp = 0.0f;
#pragma unroll
    for (int t = 0; t < 2; ++t) {
      v8f& hh = (t == 0) ? ht0 : ht1;
#pragma unroll
      for (int v = 0; v < 8; ++v) {
        const float4* cp = (const float4*)kc[t * 16 + v + hioff];
        const float4 ca = cp[0];
        const float4 cb = cp[1];
        float r = sigmoid_f(D[t][v]     + fmaf(y, ca.x, ca.w));
        float z = sigmoid_f(D[2 + t][v] + fmaf(y, ca.y, cb.x));
        float n = tanh_f(fmaf(y, ca.z, cb.y) + r * (D[4 + t][v] + cb.z));
        float hn = (1.0f - z) * n + z * hh[v];
        hh[v] = hn;
        yp = fmaf(hn, cb.w, yp);
      }
    }
    y = yp + __shfl_xor(yp, 16, 32) + bo;
    if (lo) out[col * nsteps + s] = y;
  }
}

extern "C" void kernel_launch(void* const* d_in, const int* in_sizes, int n_in,
                              void* d_out, int out_size, void* d_ws, size_t ws_size,
                              hipStream_t stream) {
  const float* in_data = (const float*)d_in[0];
  const float* hidden  = (const float*)d_in[1];
  const float* w_ih    = (const float*)d_in[2];
  const float* w_hh    = (const float*)d_in[3];
  const float* b_ih    = (const float*)d_in[4];
  const float* b_hh    = (const float*)d_in[5];
  const float* w_out   = (const float*)d_in[6];
  const float* b_out   = (const float*)d_in[7];
  const int*   nsteps  = (const int*)d_in[8];
  float* out = (float*)d_out;

  const int B = in_sizes[0] / HDIM;                            // 16384
  const int grid = (B + ROWS_PER_BLOCK - 1) / ROWS_PER_BLOCK;  // 256 blocks
  hipLaunchKernelGGL(gru_decoder_kernel, dim3(grid), dim3(WAVES_PER_BLOCK * 32),
                     0, stream,
                     in_data, hidden, w_ih, w_hh, b_ih, b_hh, w_out, b_out,
                     nsteps, out);
}
